// GCN_attn_53463752900643
// MI455X (gfx1250) — matrix operand order repaired and checked by reading, not compiled, not run on hardware
//
#include <hip/hip_runtime.h>
#include <hip/hip_bf16.h>
#include <stdint.h>
#include <stddef.h>

// ---------------------------------------------------------------------------
// GCN + dynamic attention adjacency for MI455X (gfx1250).
// All large GEMMs use V_WMMA_F32_16X16X32_BF16 (bf16 in, f32 accumulate),
// fed through LDS with double-buffered GLOBAL_LOAD_ASYNC_TO_LDS (ASYNCcnt)
// when the toolchain exposes the builtin. Each wave register-blocks a 32x32
// C tile (4 accumulators); a 256-thread block covers 128x64.
// ---------------------------------------------------------------------------

#define NN    4096
#define D_IN  256
#define EE    131072
#define OUTD  64
#define ATTD  64

typedef __attribute__((ext_vector_type(16))) __bf16 v16bf;
typedef __attribute__((ext_vector_type(8)))  float  v8f;
typedef __attribute__((ext_vector_type(4)))  int    v4i;

union Frag16 { v16bf v; unsigned u[8]; };
union Acc8   { v8f   v; float    f[8]; };

__device__ __forceinline__ unsigned short f2bf(float f) {
  unsigned u = __float_as_uint(f);
  unsigned r = u + 0x7fffu + ((u >> 16) & 1u);   // round-to-nearest-even
  return (unsigned short)(r >> 16);
}
__device__ __forceinline__ float bf2f(unsigned short h) {
  return __uint_as_float(((unsigned)h) << 16);
}

#if defined(__gfx1250__) && \
    __has_builtin(__builtin_amdgcn_global_load_async_to_lds_b128) && \
    __has_builtin(__builtin_amdgcn_s_wait_asynccnt)
#define USE_ASYNC_LDS 1
#else
#define USE_ASYNC_LDS 0
#endif

#if USE_ASYNC_LDS
typedef __attribute__((address_space(1))) v4i as1_v4i;   // global (prints as __device__)
typedef __attribute__((address_space(3))) v4i as3_v4i;   // LDS
// Flat->AS1: global VAs are identical in flat and global spaces.
// Flat->AS3: low 32 bits of a flat LDS address are the LDS byte offset
// (CDNA5 aperture rule), and AS3 pointers are 32-bit.
__device__ __forceinline__ void async_cp16(const void* g, void* l) {
  __builtin_amdgcn_global_load_async_to_lds_b128(
      (as1_v4i*)(unsigned long long)g,
      (as3_v4i*)(unsigned int)(unsigned long long)l,
      /*offset=*/0, /*cpol=*/0);
}
#endif

// ---------------------------------------------------------------------------
// WMMA GEMM: C = A * B^T  (A: MxK bf16, B: NcxK bf16, C: MxNc f32)
// mode 0: C[idx] = acc
// mode 1: Cbf[idx] = bf16(sigmoid(acc))          (C unused)
// mode 2: C[idx] = acc + bias[col]
// Requires M%128==0, Nc%64==0, K%32==0 (true for all shapes here).
// Block = 256 threads = 8 waves (4 in M x 2 in N); wave tile = 32x32.
// K pipeline: double-buffered LDS tiles (A:128x32, B:64x32), async loads
// overlap the previous tile's WMMAs.
// ---------------------------------------------------------------------------
__global__ void __launch_bounds__(256)
k_gemm_bf16(const unsigned short* __restrict__ A,
            const unsigned short* __restrict__ B,
            float* __restrict__ C,
            unsigned short* __restrict__ Cbf,
            const float* __restrict__ bias,
            int M, int Nc, int K, int mode)
{
  __shared__ unsigned short Abuf[2][128 * 32];   // 8 KB per buffer
  __shared__ unsigned short Bbuf[2][64 * 32];    // 4 KB per buffer

  const int t    = threadIdx.x;
  const int lane = t & 31;
  const int wave = t >> 5;
  const int half = lane >> 4;      // 0: lanes 0-15, 1: lanes 16-31
  const int l16  = lane & 15;
  const int ha4  = half * 4;       // A-frag K-half dword offset
  const int hb8  = half * 8;       // unused; B-frag also offsets by ha4

  const int mBase = blockIdx.y * 128;
  const int nBase = blockIdx.x * 64;
  const int wm = (wave & 3) * 32;  // wave M offset inside block tile
  const int wn = (wave >> 2) * 32; // wave N offset inside block tile

  // --- staging assignment: 16B chunks; A tile = 512 chunks (2/thread),
  //     B tile = 256 chunks (1/thread). seg = 16B segment within a 64B row.
  const int seg  = t & 3;
  const int srow = t >> 2;         // 0..63
  const unsigned short* gA0 = A + (size_t)(mBase + srow)      * (size_t)K + seg * 8;
  const unsigned short* gA1 = A + (size_t)(mBase + 64 + srow) * (size_t)K + seg * 8;
  const unsigned short* gB0 = B + (size_t)(nBase + srow)      * (size_t)K + seg * 8;
  const int lOffA0 = srow * 32 + seg * 8;
  const int lOffA1 = (64 + srow) * 32 + seg * 8;
  const int lOffB0 = srow * 32 + seg * 8;

  auto stage = [&](int k0, int buf) {
#if USE_ASYNC_LDS
    async_cp16(gA0 + k0, &Abuf[buf][lOffA0]);
    async_cp16(gA1 + k0, &Abuf[buf][lOffA1]);
    async_cp16(gB0 + k0, &Bbuf[buf][lOffB0]);
#else
    *(uint4*)&Abuf[buf][lOffA0] = *(const uint4*)(gA0 + k0);
    *(uint4*)&Abuf[buf][lOffA1] = *(const uint4*)(gA1 + k0);
    *(uint4*)&Bbuf[buf][lOffB0] = *(const uint4*)(gB0 + k0);
#endif
  };

  Acc8 acc[2][2];
#pragma unroll
  for (int i = 0; i < 2; ++i)
#pragma unroll
    for (int j = 0; j < 2; ++j)
#pragma unroll
      for (int e = 0; e < 8; ++e) acc[i][j].f[e] = 0.0f;

  const int nk = K >> 5;
  stage(0, 0);

  for (int kt = 0; kt < nk; ++kt) {
    const int buf = kt & 1;
#if USE_ASYNC_LDS
    __builtin_amdgcn_s_wait_asynccnt(0);
#endif
    __syncthreads();                       // tile `buf` resident for all waves
    if (kt + 1 < nk) stage((kt + 1) << 5, buf ^ 1);   // overlap with compute

    // --- fragment gathers from LDS (ISA 7.12.2 layouts) ---
    const unsigned* pa0 = (const unsigned*)&Abuf[buf][(wm + l16) * 32];
    const unsigned* pa1 = (const unsigned*)&Abuf[buf][(wm + 16 + l16) * 32];
    const unsigned* pb0 = (const unsigned*)&Bbuf[buf][(wn + l16) * 32];
    const unsigned* pb1 = (const unsigned*)&Bbuf[buf][(wn + 16 + l16) * 32];
    Frag16 fa[2], fb[2];
#pragma unroll
    for (int j = 0; j < 4; ++j) {          // A: V0..3 -> K{0..7}, V4..7 -> K{16..23} (+half*8)
      fa[0].u[j]     = pa0[j + ha4];
      fa[0].u[j + 4] = pa0[j + 8 + ha4];
      fa[1].u[j]     = pa1[j + ha4];
      fa[1].u[j + 4] = pa1[j + 8 + ha4];
    }
#pragma unroll
    for (int v = 0; v < 8; ++v) {          // B: V0..3 -> K{0..7}, V4..7 -> K{16..23} (+half*8)
      fb[0].u[v] = pb0[(v & 3) + ((v >> 2) << 3) + ha4];
      fb[1].u[v] = pb1[(v & 3) + ((v >> 2) << 3) + ha4];
    }

    acc[0][0].v = __builtin_amdgcn_wmma_f32_16x16x32_bf16(
        false, fa[0].v, false, fb[0].v, (short)0, acc[0][0].v, false, false);
    acc[0][1].v = __builtin_amdgcn_wmma_f32_16x16x32_bf16(
        false, fa[0].v, false, fb[1].v, (short)0, acc[0][1].v, false, false);
    acc[1][0].v = __builtin_amdgcn_wmma_f32_16x16x32_bf16(
        false, fa[1].v, false, fb[0].v, (short)0, acc[1][0].v, false, false);
    acc[1][1].v = __builtin_amdgcn_wmma_f32_16x16x32_bf16(
        false, fa[1].v, false, fb[1].v, (short)0, acc[1][1].v, false, false);

    __syncthreads();                       // everyone done reading `buf`
  }

  // C/D layout: VGPR i -> row +(i + 8*half), lane%16 -> column.
#pragma unroll
  for (int si = 0; si < 2; ++si) {
#pragma unroll
    for (int sj = 0; sj < 2; ++sj) {
      const int col = nBase + wn + sj * 16 + l16;
#pragma unroll
      for (int i = 0; i < 8; ++i) {
        const int row = mBase + wm + si * 16 + half * 8 + i;
        const size_t idx = (size_t)row * (size_t)Nc + (size_t)col;
        const float v = acc[si][sj].f[i];
        if (mode == 0) {
          C[idx] = v;
        } else if (mode == 1) {
          Cbf[idx] = f2bf(1.0f / (1.0f + expf(-v)));
        } else {
          C[idx] = v + bias[col];
        }
      }
    }
  }
}

// ---------------------------------------------------------------------------
// Elementwise / reduction helpers
// ---------------------------------------------------------------------------
__global__ void k_zero_f32(float* p, long n) {
  long i = (long)blockIdx.x * blockDim.x + threadIdx.x;
  long s = (long)gridDim.x * blockDim.x;
  for (; i < n; i += s) p[i] = 0.0f;
}

__global__ void k_scatter(const int* __restrict__ ei,
                          const float* __restrict__ ew,
                          float* __restrict__ Adense) {
  int e = blockIdx.x * blockDim.x + threadIdx.x;
  if (e < EE)
    atomicAdd(&Adense[(size_t)ei[e] * NN + (size_t)ei[EE + e]], ew[e]);
}

__global__ void k_cvt_bf16(const float* __restrict__ s,
                           unsigned short* __restrict__ d, long n) {
  long i = (long)blockIdx.x * blockDim.x + threadIdx.x;
  long st = (long)gridDim.x * blockDim.x;
  for (; i < n; i += st) d[i] = f2bf(s[i]);
}

// dst[c*R + r] = bf16(src[r*C + c])   (transpose + convert)
__global__ void k_tcvt_bf16(const float* __restrict__ s,
                            unsigned short* __restrict__ d, int R, int Cc) {
  long i = (long)blockIdx.x * blockDim.x + threadIdx.x;
  if (i >= (long)R * Cc) return;
  int r = (int)(i / Cc), c = (int)(i % Cc);
  d[(size_t)c * R + r] = f2bf(s[i]);
}

// Additive attention over the 2 adjacency candidates; one thread per node row.
__global__ void k_attn(const float* __restrict__ T,   // [2][NN][ATTD]
                       const float* __restrict__ ab1, // [ATTD]
                       const float* __restrict__ aw2, // [ATTD]
                       const float* __restrict__ ab2, // [1]
                       float* __restrict__ a01)       // [NN][2]
{
  int r = blockIdx.x * blockDim.x + threadIdx.x;
  if (r >= NN) return;
  float s[2];
  for (int c = 0; c < 2; ++c) {
    const float* Tr = T + (size_t)c * NN * ATTD + (size_t)r * ATTD;
    float acc = ab2[0];
    for (int j = 0; j < ATTD; ++j) acc += tanhf(Tr[j] + ab1[j]) * aw2[j];
    s[c] = acc;
  }
  float m = fmaxf(s[0], s[1]);
  float e0 = expf(s[0] - m), e1 = expf(s[1] - m);
  float inv = 1.0f / (e0 + e1);
  a01[2 * r]     = e0 * inv;
  a01[2 * r + 1] = e1 * inv;
}

// A_m[r,:] = a0*A_orig + a1*A_star (bf16 in, f32 out) and deg[r] = rowsum + 1
__global__ void __launch_bounds__(256)
k_blend_deg(const unsigned short* __restrict__ Ao,
            const unsigned short* __restrict__ As,
            const float* __restrict__ a01,
            float* __restrict__ Am, float* __restrict__ deg) {
  __shared__ float red[256];
  const int r = blockIdx.x, t = threadIdx.x;
  const float a0 = a01[2 * r], a1 = a01[2 * r + 1];
  const size_t base = (size_t)r * NN;
  float s = 0.0f;
  for (int c = t; c < NN; c += 256) {
    float v = a0 * bf2f(Ao[base + c]) + a1 * bf2f(As[base + c]);
    Am[base + c] = v;
    s += v;
  }
  red[t] = s; __syncthreads();
  for (int o = 128; o > 0; o >>= 1) {
    if (t < o) red[t] += red[t + o];
    __syncthreads();
  }
  if (t == 0) deg[r] = red[0] + 1.0f;   // + self loop
}

__global__ void k_dinv(float* deg) {
  int r = blockIdx.x * blockDim.x + threadIdx.x;
  if (r >= NN) return;
  float d = deg[r];
  deg[r] = (d > 0.0f) ? rsqrtf(d) : 0.0f;
}

// An_bf[r,c] = bf16(dinv[r] * (A_m + I)[r,c] * dinv[c])
__global__ void k_norm_adj(const float* __restrict__ Am,
                           const float* __restrict__ dinv,
                           unsigned short* __restrict__ Anb) {
  size_t i = (size_t)blockIdx.x * blockDim.x + threadIdx.x;
  size_t st = (size_t)gridDim.x * blockDim.x;
  const size_t total = (size_t)NN * NN;
  for (; i < total; i += st) {
    int r = (int)(i >> 12), c = (int)(i & (NN - 1));
    float v = Am[i] + ((r == c) ? 1.0f : 0.0f);
    Anb[i] = f2bf(v * dinv[r] * dinv[c]);
  }
}

// BatchNorm stats over nodes: one block per channel.
__global__ void __launch_bounds__(256)
k_bn_stats(const float* __restrict__ H, int dout,
           float* __restrict__ mu, float* __restrict__ rstd) {
  __shared__ float s1[256], s2[256];
  const int j = blockIdx.x, t = threadIdx.x;
  float a = 0.0f, b = 0.0f;
  for (int r = t; r < NN; r += 256) {
    float x = H[(size_t)r * dout + j];
    a += x; b += x * x;
  }
  s1[t] = a; s2[t] = b; __syncthreads();
  for (int o = 128; o > 0; o >>= 1) {
    if (t < o) { s1[t] += s1[t + o]; s2[t] += s2[t + o]; }
    __syncthreads();
  }
  if (t == 0) {
    float m = s1[0] / (float)NN;
    float v = s2[0] / (float)NN - m * m;   // biased var (jnp default)
    mu[j] = m;
    rstd[j] = rsqrtf(v + 1e-5f);
  }
}

// BN apply + LeakyReLU(0.01) -> next-layer h in bf16.
__global__ void k_bn_apply(const float* __restrict__ H,
                           const float* __restrict__ mu,
                           const float* __restrict__ rstd,
                           const float* __restrict__ g,
                           const float* __restrict__ b,
                           unsigned short* __restrict__ hb, int dout) {
  long i = (long)blockIdx.x * blockDim.x + threadIdx.x;
  if (i >= (long)NN * dout) return;
  int j = (int)(i % dout);
  float y = g[j] * (H[i] - mu[j]) * rstd[j] + b[j];
  y = (y > 0.0f) ? y : 0.01f * y;
  hb[i] = f2bf(y);
}

// ---------------------------------------------------------------------------
// Host orchestration
// ---------------------------------------------------------------------------
static inline size_t alignup(size_t x) { return (x + 255) & ~(size_t)255; }

extern "C" void kernel_launch(void* const* d_in, const int* in_sizes, int n_in,
                              void* d_out, int out_size, void* d_ws, size_t ws_size,
                              hipStream_t stream) {
  (void)in_sizes; (void)n_in; (void)out_size; (void)ws_size;

  const float* X  = (const float*)d_in[0];
  const int*   ei = (const int*)d_in[1];
  const float* ew = (const float*)d_in[2];
  const float* lin_w = (const float*)d_in[27];
  const float* lin_b = (const float*)d_in[28];

  float* out_ptr = (float*)d_out;                    // [NN, OUTD]
  float* Am      = out_ptr + (size_t)NN * OUTD;      // final_A region, reused as
                                                     // f32 A scratch every layer

  // ---- workspace carve-up ----
  char* ws = (char*)d_ws; size_t o = 0;
  unsigned short* Aorig_bf = (unsigned short*)(ws + o); o += alignup((size_t)NN * NN * 2);
  unsigned short* Astar_bf = (unsigned short*)(ws + o); o += alignup((size_t)NN * NN * 2);
  unsigned short* An_bf    = (unsigned short*)(ws + o); o += alignup((size_t)NN * NN * 2);
  unsigned short* h_bf     = (unsigned short*)(ws + o); o += alignup((size_t)NN * 256 * 2);
  unsigned short* aw1T     = (unsigned short*)(ws + o); o += alignup((size_t)ATTD * NN * 2);
  unsigned short* wT       = (unsigned short*)(ws + o); o += alignup((size_t)256 * 256 * 2);
  unsigned short* PT       = (unsigned short*)(ws + o); o += alignup((size_t)256 * NN * 2);
  unsigned short* linT     = (unsigned short*)(ws + o); o += alignup((size_t)OUTD * 128 * 2);
  float* P    = (float*)(ws + o); o += alignup((size_t)NN * 256 * 4);
  float* Hp   = (float*)(ws + o); o += alignup((size_t)NN * 256 * 4);
  float* T    = (float*)(ws + o); o += alignup((size_t)2 * NN * ATTD * 4);
  float* a01  = (float*)(ws + o); o += alignup((size_t)NN * 2 * 4);
  float* deg  = (float*)(ws + o); o += alignup((size_t)NN * 4);
  float* mu   = (float*)(ws + o); o += alignup((size_t)256 * 4);
  float* rstd = (float*)(ws + o); o += alignup((size_t)256 * 4);

  const int dims[4] = {D_IN, 256, 256, 128};

  // ---- A_orig: scatter edges into dense f32 (scratch = final_A region),
  //      then convert once to bf16 for the attention GEMM operand.
  k_zero_f32<<<4096, 256, 0, stream>>>(Am, (long)NN * NN);
  k_scatter<<<(EE + 255) / 256, 256, 0, stream>>>(ei, ew, Am);
  k_cvt_bf16<<<4096, 256, 0, stream>>>(Am, Aorig_bf, (long)NN * NN);

  // h0 = X in bf16
  k_cvt_bf16<<<4096, 256, 0, stream>>>(X, h_bf, (long)NN * D_IN);

  for (int i = 0; i < 3; ++i) {
    const int din = dims[i], dout = dims[i + 1];
    const int L = 3 + 8 * i;
    const float* gcn_w = (const float*)d_in[L + 0];
    const float* gcn_b = (const float*)d_in[L + 1];
    const float* bn_g  = (const float*)d_in[L + 2];
    const float* bn_b  = (const float*)d_in[L + 3];
    const float* at_w1 = (const float*)d_in[L + 4];
    const float* at_b1 = (const float*)d_in[L + 5];
    const float* at_w2 = (const float*)d_in[L + 6];
    const float* at_b2 = (const float*)d_in[L + 7];

    // A_star = sigmoid(h @ h^T) -> bf16   (M=Nc=4096, K=din)
    k_gemm_bf16<<<dim3(NN / 64, NN / 128), 256, 0, stream>>>(
        h_bf, h_bf, nullptr, Astar_bf, nullptr, NN, NN, din, 1);

    // attention candidate scores: T_c = A_c @ aw1   (Nc=ATTD, K=4096)
    k_tcvt_bf16<<<((long)NN * ATTD + 255) / 256, 256, 0, stream>>>(at_w1, aw1T, NN, ATTD);
    k_gemm_bf16<<<dim3(ATTD / 64, NN / 128), 256, 0, stream>>>(
        Aorig_bf, aw1T, T, nullptr, nullptr, NN, ATTD, NN, 0);
    k_gemm_bf16<<<dim3(ATTD / 64, NN / 128), 256, 0, stream>>>(
        Astar_bf, aw1T, T + (size_t)NN * ATTD, nullptr, nullptr, NN, ATTD, NN, 0);

    // per-row additive attention softmax over the 2 candidates
    k_attn<<<(NN + 255) / 256, 256, 0, stream>>>(T, at_b1, at_w2, at_b2, a01);

    // A_m = a0*A_orig + a1*A_star (f32, written into d_out final_A region),
    // deg = rowsum(A_m) + 1
    k_blend_deg<<<NN, 256, 0, stream>>>(Aorig_bf, Astar_bf, a01, Am, deg);
    k_dinv<<<(NN + 255) / 256, 256, 0, stream>>>(deg);

    // An = D^-1/2 (A_m + I) D^-1/2 -> bf16
    k_norm_adj<<<8192, 256, 0, stream>>>(Am, deg, An_bf);

    // P = h @ gcn_w   (Nc=dout, K=din)
    k_tcvt_bf16<<<((long)din * dout + 255) / 256, 256, 0, stream>>>(gcn_w, wT, din, dout);
    k_gemm_bf16<<<dim3(dout / 64, NN / 128), 256, 0, stream>>>(
        h_bf, wT, P, nullptr, nullptr, NN, dout, din, 0);

    // Hp = An @ P + gcn_b   (Nc=dout, K=4096)
    k_tcvt_bf16<<<((long)NN * dout + 255) / 256, 256, 0, stream>>>(P, PT, NN, dout);
    k_gemm_bf16<<<dim3(dout / 64, NN / 128), 256, 0, stream>>>(
        An_bf, PT, Hp, nullptr, gcn_b, NN, dout, NN, 2);

    // BatchNorm (training-mode batch stats) + LeakyReLU -> next h (bf16)
    k_bn_stats<<<dout, 256, 0, stream>>>(Hp, dout, mu, rstd);
    k_bn_apply<<<((long)NN * dout + 255) / 256, 256, 0, stream>>>(
        Hp, mu, rstd, bn_g, bn_b, h_bf, dout);
  }

  // out = h @ lin_w + lin_b   (M=4096, Nc=64, K=128)
  k_tcvt_bf16<<<((long)128 * OUTD + 255) / 256, 256, 0, stream>>>(lin_w, linT, 128, OUTD);
  k_gemm_bf16<<<dim3(OUTD / 64, NN / 128), 256, 0, stream>>>(
      h_bf, linT, out_ptr, nullptr, lin_b, NN, OUTD, 128, 2);
}
